// DU_Core_V1_52888227283304
// MI455X (gfx1250) — matrix operand order repaired and checked by reading, not compiled, not run on hardware
//
#include <hip/hip_runtime.h>

typedef _Float16 v8h  __attribute__((ext_vector_type(8)));
typedef _Float16 v16h __attribute__((ext_vector_type(16)));
typedef float    v8f  __attribute__((ext_vector_type(8)));

#define BETA    0.95f
#define THRESH  1.0f
#define T_STEPS 128
#define BATCH   32

static __device__ __forceinline__ v16h cat8(v8h lo, v8h hi) {
    return __builtin_shufflevector(lo, hi, 0,1,2,3,4,5,6,7,8,9,10,11,12,13,14,15);
}

// ---------------------------------------------------------------------------
// One-time per launch: w[N,K] f32  ->  hiT/loT [N,K] f16 split (hi + residual)
// Not transposed: B-fragments (N-column per lane, K contiguous) load as b128s.
// ---------------------------------------------------------------------------
__global__ void cvt_weights(const float* __restrict__ w,
                            _Float16* __restrict__ hiT,
                            _Float16* __restrict__ loT, int K, int N) {
    size_t i = (size_t)blockIdx.x * blockDim.x + threadIdx.x;
    size_t total = (size_t)K * N;
    if (i >= total) return;
    int n = (int)(i / K);          // coalesced read of w
    int k = (int)(i % K);
    float f = w[i];
    _Float16 h = (_Float16)f;
    float r = f - (float)h;
    hiT[(size_t)n * K + k] = h;
    loT[(size_t)n * K + k] = (_Float16)r;
}

// One-time per launch: all input spike frames f32 -> f16 (spikes are 0/1,
// exact in f16) so the recurrent path never converts.
__global__ void cvt_input(const float* __restrict__ x, _Float16* __restrict__ xh,
                          size_t total) {
    size_t i = (size_t)blockIdx.x * blockDim.x + threadIdx.x;
    if (i < total) xh[i] = (_Float16)x[i];
}

__global__ void zero_f32(float* __restrict__ p, int n) {
    int i = blockIdx.x * blockDim.x + threadIdx.x;
    if (i < n) p[i] = 0.0f;
}

// ---------------------------------------------------------------------------
// One (timestep, layer) step: syn = x @ W.T + b ; LIF update ; emit spikes.
// Block = 256 threads (8 waves) owns ONE 16x16 output tile, split-K by 8:
// each wave runs v_wmma_f32_16x16x32_f16 over its K/8 slice (hi + lo weight
// splits, f32 accumulate), partials reduced in fixed order through LDS,
// then 1 element/thread leaky-integrate-fire on the membrane state.
// K, N compile-time so the unrolled load stream uses pure immediate offsets.
// ---------------------------------------------------------------------------
template<int K, int N>
__global__ __launch_bounds__(256)
void snn_layer(const _Float16* __restrict__ xin,
               const _Float16* __restrict__ whiT,
               const _Float16* __restrict__ wloT,
               const float* __restrict__ bias,
               float* __restrict__ mem,
               _Float16* __restrict__ act,      // nullptr on last layer
               float* __restrict__ spk_f32) {   // nullptr except last layer
    __shared__ float lds_part[8][256];          // [kslice][m*16 + n]

    constexpr int NT = N / 16;                  // n-tiles
    constexpr int KS = K / 8;                   // K per wave slice

    const int tid    = threadIdx.x;
    const int lane   = tid & 31;
    const int kslice = tid >> 5;                // 0..7
    const int ntile  = blockIdx.x % NT;
    const int mtile  = blockIdx.x / NT;         // 0..1
    const int n_lane = lane & 15;
    const int hi16   = lane >> 4;               // 0/1

    // Accumulator: bias folded into the kslice==0 partial (n fixed per lane).
    v8f c;
    {
        float binit = (kslice == 0) ? bias[ntile * 16 + n_lane] : 0.0f;
        #pragma unroll
        for (int r = 0; r < 8; ++r) c[r] = binit;
    }

    const int k0   = kslice * KS;
    const int arow = mtile * 16 + n_lane;       // A-matrix row for this lane

    // Per-wave base pointers; all k-loop displacements are compile-time.
    const _Float16* xp = xin + (size_t)arow * K + k0 + hi16 * 8;
    const _Float16* bp = whiT + (size_t)(ntile * 16 + n_lane) * K + k0 + hi16 * 8;
    const _Float16* lp = wloT + (size_t)(ntile * 16 + n_lane) * K + k0 + hi16 * 8;

    #pragma unroll
    for (int kk = 0; kk < KS; kk += 32) {
        // ---- A fragment: lane holds row m, K chunks (base, base+16) ----
        v16h a = cat8(*(const v8h*)(xp + kk), *(const v8h*)(xp + kk + 16));

        // ---- B fragments: lane holds column n, K chunks (base, base+16) ----
        const _Float16* bpp = bp + kk;
        v16h bh = cat8(*(const v8h*)bpp, *(const v8h*)(bpp + 16));
        const _Float16* lpp = lp + kk;
        v16h bl = cat8(*(const v8h*)lpp, *(const v8h*)(lpp + 16));

        // split-f16 GEMM: A x (w_hi + w_lo), f32 accumulate
        c = __builtin_amdgcn_wmma_f32_16x16x32_f16(false, a, false, bh,
                                                   (short)0, c, false, false);
        c = __builtin_amdgcn_wmma_f32_16x16x32_f16(false, a, false, bl,
                                                   (short)0, c, false, false);
    }

    // ---- deterministic split-K reduction through LDS ----
    {
        const int base = (hi16 * 8) * 16 + n_lane;
        #pragma unroll
        for (int r = 0; r < 8; ++r)
            lds_part[kslice][base + r * 16] = c[r];
    }
    __syncthreads();

    // ---- LIF elementwise: one tile element per thread ----
    {
        const int e  = tid;                     // m*16 + n
        const int mg = mtile * 16 + (e >> 4);
        const int ng = ntile * 16 + (e & 15);
        float syn = 0.0f;
        #pragma unroll
        for (int s8 = 0; s8 < 8; ++s8) syn += lds_part[s8][e];
        const size_t gi = (size_t)mg * N + ng;
        const float mv = BETA * mem[gi] + syn;          // leaky integrate
        const float s  = (mv > THRESH) ? 1.0f : 0.0f;   // fire
        mem[gi] = mv - s * THRESH;                      // reset-by-subtract
        if (act)     act[gi]     = (_Float16)s;
        if (spk_f32) spk_f32[gi] = s;
    }
}

// ---------------------------------------------------------------------------
extern "C" void kernel_launch(void* const* d_in, const int* in_sizes, int n_in,
                              void* d_out, int out_size, void* d_ws, size_t ws_size,
                              hipStream_t stream) {
    const float* x_in = (const float*)d_in[0];   // [128, 32, 1024] f32 spikes
    const float* w0   = (const float*)d_in[1];   // [2048, 1024]
    const float* b0   = (const float*)d_in[2];
    const float* w1   = (const float*)d_in[3];   // [2048, 2048]
    const float* b1   = (const float*)d_in[4];
    const float* w2   = (const float*)d_in[5];   // [1024, 2048]
    const float* b2   = (const float*)d_in[6];
    float* out = (float*)d_out;                  // [128, 32, 1024]

    // ---- workspace carve-up (all 256B aligned) ----
    char* ws = (char*)d_ws;
    size_t off = 0;
    auto carve = [&](size_t bytes) -> void* {
        void* p = ws + off;
        off += (bytes + 255) & ~(size_t)255;
        return p;
    };
    _Float16* whi0 = (_Float16*)carve((size_t)1024 * 2048 * 2);
    _Float16* wlo0 = (_Float16*)carve((size_t)1024 * 2048 * 2);
    _Float16* whi1 = (_Float16*)carve((size_t)2048 * 2048 * 2);
    _Float16* wlo1 = (_Float16*)carve((size_t)2048 * 2048 * 2);
    _Float16* whi2 = (_Float16*)carve((size_t)2048 * 1024 * 2);
    _Float16* wlo2 = (_Float16*)carve((size_t)2048 * 1024 * 2);
    _Float16* xh   = (_Float16*)carve((size_t)T_STEPS * BATCH * 1024 * 2);
    float* mem0 = (float*)carve((size_t)BATCH * 2048 * 4);
    float* mem1 = (float*)carve((size_t)BATCH * 2048 * 4);
    float* mem2 = (float*)carve((size_t)BATCH * 1024 * 4);
    _Float16* act1 = (_Float16*)carve((size_t)BATCH * 2048 * 2);
    _Float16* act2 = (_Float16*)carve((size_t)BATCH * 2048 * 2);
    (void)ws_size; (void)in_sizes; (void)n_in; (void)out_size;

    // ---- one-time (per launch) conversions; zero membranes ----
    {
        size_t n;
        n = (size_t)1024 * 2048;
        cvt_weights<<<(unsigned)((n + 255) / 256), 256, 0, stream>>>(w0, whi0, wlo0, 1024, 2048);
        n = (size_t)2048 * 2048;
        cvt_weights<<<(unsigned)((n + 255) / 256), 256, 0, stream>>>(w1, whi1, wlo1, 2048, 2048);
        n = (size_t)2048 * 1024;
        cvt_weights<<<(unsigned)((n + 255) / 256), 256, 0, stream>>>(w2, whi2, wlo2, 2048, 1024);
        n = (size_t)T_STEPS * BATCH * 1024;
        cvt_input<<<(unsigned)((n + 255) / 256), 256, 0, stream>>>(x_in, xh, n);
        int ztot = BATCH * (2048 + 2048 + 1024);   // mem0..mem2 are contiguous
        zero_f32<<<(ztot + 255) / 256, 256, 0, stream>>>(mem0, ztot);
    }

    // ---- the recurrence: 128 timesteps x 3 layers ----
    for (int t = 0; t < T_STEPS; ++t) {
        const _Float16* xt = xh + (size_t)t * BATCH * 1024;
        snn_layer<1024, 2048><<<2 * (2048 / 16), 256, 0, stream>>>(
            xt,   whi0, wlo0, b0, mem0, act1, (float*)nullptr);
        snn_layer<2048, 2048><<<2 * (2048 / 16), 256, 0, stream>>>(
            act1, whi1, wlo1, b1, mem1, act2, (float*)nullptr);
        snn_layer<2048, 1024><<<2 * (1024 / 16), 256, 0, stream>>>(
            act2, whi2, wlo2, b2, mem2, (_Float16*)nullptr,
            out + (size_t)t * BATCH * 1024);
    }
}
